// MHLPPredictor_23424751633139
// MI455X (gfx1250) — matrix operand order repaired and checked by reading, not compiled, not run on hardware
//
#include <hip/hip_runtime.h>

typedef _Float16 half_t;
typedef __attribute__((ext_vector_type(16))) _Float16 v16h;
typedef __attribute__((ext_vector_type(8)))  float    v8f;

#define TM 16          // samples per wave-tile
#define NWAVES 8
#define NTHREADS 256   // 8 waves (wave32); 128 samples per block

// ---------------- dynamic LDS layout ----------------
// shared weights (f16) + constants:
//   [     0, 24576)  sWin  f16 192x64   (in_proj_w)
//   [ 24576, 40960)  sW2   f16  64x128
//   [ 40960, 49152)  sWout f16  64x64
//   [ 49152, 53248)  sW3   f16  32x64
//   [ 53248, 57344)  small constants (biases, LN params, W4, b4)
// per-wave scratch (8 x 20992 bytes), time-aliased:
//   +0     (4096)  region A: wHA f16 16x128 -> wCtx f16 32x64 -> wPool f16 16x64
//   +4096  (4096)  region B: wComb f16 32x64 -> wHead f32 16x32
//   +8192  (12288) region C: wTmp f32 16x64 -> wQKV f16 32x192 -> wXf f32 32x64
//   +20480 (384)   wCols int[80], wHW int[16]
#define OFF_WIN   0
#define OFF_W2    24576
#define OFF_WOUT  40960
#define OFF_W3    49152
#define OFF_B1    53248
#define OFF_B2    53760
#define OFF_BIN   54016
#define OFF_BOUT  54784
#define OFF_B3    55040
#define OFF_L1G   55168
#define OFF_L1B   55424
#define OFF_L2G   55680
#define OFF_L2B   55936
#define OFF_W4    56192
#define OFF_B4    56320
#define WAVE_BASE   57344
#define WAVE_STRIDE 20992
#define SMEM_TOTAL  (WAVE_BASE + NWAVES * WAVE_STRIDE)   // 225280 B <= 320KB WGP LDS

#define WAVE_SYNC() __builtin_amdgcn_wave_barrier()

__device__ inline v8f wmma_f16(v16h a, v16h b, v8f c) {
  return __builtin_amdgcn_wmma_f32_16x16x32_f16(false, a, false, b, (short)0, c,
                                                false, false);
}

// A fragment (16x32 f16) from row-major f16 [16][ldk] at column k0.
__device__ inline v16h load_a(const half_t* base, int ldk, int k0, int lane) {
  int row = lane & 15;
  int sel = lane >> 4;
  const half_t* p = base + row * ldk + k0 + sel * 8;
  union { uint4 u[2]; v16h v; } t;
  t.u[0] = *(const uint4*)(p);
  t.u[1] = *(const uint4*)(p + 16);
  return t.v;
}

// B fragment (32x16 f16), B(k,n) = W[n][k], W row-major f16 [N][ldk], base at row n0.
__device__ inline v16h load_b(const half_t* base_n0, int ldk, int k0, int lane) {
  int n  = lane & 15;
  int kh = (lane >> 4) * 8;
  const half_t* p = base_n0 + n * ldk + k0 + kh;
  union { uint4 u[2]; v16h v; } t;
  t.u[0] = *(const uint4*)(p);
  t.u[1] = *(const uint4*)(p + 16);
  return t.v;
}

__global__ __launch_bounds__(NTHREADS)
void mhlp_predictor_kernel(
    const int* __restrict__ hw_idx, const int* __restrict__ op_idx,
    const int* __restrict__ wd_idx, const float* __restrict__ hw_embed,
    const float* __restrict__ W1, const float* __restrict__ b1,
    const float* __restrict__ W2, const float* __restrict__ b2,
    const float* __restrict__ ln1g, const float* __restrict__ ln1b,
    const float* __restrict__ Win, const float* __restrict__ bin,
    const float* __restrict__ Wout, const float* __restrict__ bout,
    const float* __restrict__ ln2g, const float* __restrict__ ln2b,
    const float* __restrict__ W3, const float* __restrict__ b3,
    const float* __restrict__ W4, const float* __restrict__ b4,
    float* __restrict__ out, int B) {
  extern __shared__ __align__(16) unsigned char smem[];

  half_t* sWin  = (half_t*)(smem + OFF_WIN);
  half_t* sW2   = (half_t*)(smem + OFF_W2);
  half_t* sWout = (half_t*)(smem + OFF_WOUT);
  half_t* sW3   = (half_t*)(smem + OFF_W3);
  float*  sB1   = (float*)(smem + OFF_B1);
  float*  sB2   = (float*)(smem + OFF_B2);
  float*  sBin  = (float*)(smem + OFF_BIN);
  float*  sBout = (float*)(smem + OFF_BOUT);
  float*  sB3   = (float*)(smem + OFF_B3);
  float*  sL1g  = (float*)(smem + OFF_L1G);
  float*  sL1b  = (float*)(smem + OFF_L1B);
  float*  sL2g  = (float*)(smem + OFF_L2G);
  float*  sL2b  = (float*)(smem + OFF_L2B);
  float*  sW4   = (float*)(smem + OFF_W4);
  float*  sB4   = (float*)(smem + OFF_B4);

  const int t    = threadIdx.x;
  const int lane = t & 31;
  const int w    = t >> 5;
  const int g0   = blockIdx.x * (NWAVES * TM) + w * TM;  // this wave's 16 samples

  unsigned char* wbase = smem + WAVE_BASE + w * WAVE_STRIDE;
  half_t* wHA   = (half_t*)(wbase + 0);       // 16x128
  half_t* wCtx  = (half_t*)(wbase + 0);       // 32x64  (after wHA dead)
  half_t* wPool = (half_t*)(wbase + 0);       // 16x64  (after wCtx dead)
  half_t* wComb = (half_t*)(wbase + 4096);    // 32x64: [tok][m][d]
  float*  wHead = (float*)(wbase + 4096);     // 16x32  (after wComb dead)
  float*  wTmp  = (float*)(wbase + 8192);     // 16x64
  half_t* wQKV  = (half_t*)(wbase + 8192);    // 32x192 (after wTmp dead)
  float*  wXf   = (float*)(wbase + 8192);     // 32x64  (after wQKV dead)
  int*    wCols = (int*)(wbase + 20480);      // 16*5
  int*    wHW   = (int*)(wbase + 20800);      // 16

  // ---- stage 0 (block-wide, once): stage all weights as f16 + constants ----
  for (int e = t; e < 192 * 64; e += NTHREADS) sWin[e]  = (half_t)Win[e];
  for (int e = t; e < 64 * 128; e += NTHREADS) sW2[e]   = (half_t)W2[e];
  for (int e = t; e < 64 * 64;  e += NTHREADS) sWout[e] = (half_t)Wout[e];
  for (int e = t; e < 32 * 64;  e += NTHREADS) sW3[e]   = (half_t)W3[e];
  if (t < 128) sB1[t] = b1[t];
  if (t < 64) {
    sB2[t] = b2[t]; sBout[t] = bout[t];
    sL1g[t] = ln1g[t]; sL1b[t] = ln1b[t];
    sL2g[t] = ln2g[t]; sL2b[t] = ln2b[t];
  }
  if (t < 32) { sB3[t] = b3[t]; sW4[t] = W4[t]; }
  for (int e = t; e < 192; e += NTHREADS) sBin[e] = bin[e];
  if (t == 0) sB4[0] = b4[0];
  // per-wave sample indices (one sample per lane<16)
  if (lane < TM) {
    int s = g0 + lane; if (s >= B) s = B - 1;
    wHW[lane] = hw_idx[s];
#pragma unroll
    for (int l = 0; l < 5; ++l)
      wCols[lane * 5 + l] = l * 15 + op_idx[s * 5 + l] * 3 + wd_idx[s * 5 + l];
  }
  __syncthreads();   // the ONLY block-wide barrier; everything below is wave-private

  // ---- stage 1: token 0 (hw embedding) + gather-sum h = relu(onehot@W1^T+b1) ----
  for (int e = lane; e < TM * 64; e += 32) {
    int m = e >> 6, d = e & 63;
    wComb[m * 64 + d] = (half_t)hw_embed[wHW[m] * 64 + d];
  }
#pragma unroll
  for (int jj = 0; jj < 4; ++jj) {
    const int j = lane + 32 * jj;            // 128 hidden units over 32 lanes
    const float* w1r = W1 + j * 75;
    const float bj = sB1[j];
    for (int m = 0; m < TM; ++m) {
      float acc = bj;
#pragma unroll
      for (int l = 0; l < 5; ++l) acc += w1r[wCols[m * 5 + l]];
      wHA[m * 128 + j] = (half_t)fmaxf(acc, 0.0f);
    }
  }
  WAVE_SYNC();

  // ---- stage 2: arch_pre = h @ W2^T + b2 (16x128x64), double-buffered B ----
  {
    v16h a0 = load_a(wHA, 128, 0,  lane);
    v16h a1 = load_a(wHA, 128, 32, lane);
    v16h a2 = load_a(wHA, 128, 64, lane);
    v16h a3 = load_a(wHA, 128, 96, lane);
    const int col = lane & 15, sel = lane >> 4;
    float bias[4];
#pragma unroll
    for (int nt = 0; nt < 4; ++nt) bias[nt] = sB2[nt * 16 + col];
    v16h b0 = load_b(sW2, 128, 0,  lane);
    v16h b1 = load_b(sW2, 128, 32, lane);
    v16h b2v = load_b(sW2, 128, 64, lane);
    v16h b3v = load_b(sW2, 128, 96, lane);
#pragma unroll
    for (int nt = 0; nt < 4; ++nt) {
      const int n0 = nt * 16;
      v16h nb0, nb1, nb2, nb3;
      if (nt < 3) {
        const half_t* nb = sW2 + (n0 + 16) * 128;
        nb0 = load_b(nb, 128, 0,  lane);
        nb1 = load_b(nb, 128, 32, lane);
        nb2 = load_b(nb, 128, 64, lane);
        nb3 = load_b(nb, 128, 96, lane);
      }
      v8f c = {};
      c = wmma_f16(a0, b0,  c);
      c = wmma_f16(a1, b1,  c);
      c = wmma_f16(a2, b2v, c);
      c = wmma_f16(a3, b3v, c);
#pragma unroll
      for (int r = 0; r < 8; ++r)
        wTmp[(r + sel * 8) * 64 + n0 + col] = c[r] + bias[nt];
      b0 = nb0; b1 = nb1; b2v = nb2; b3v = nb3;
    }
  }
  WAVE_SYNC();
  // LayerNorm1 -> token 1 of combined (lane<16, one sample per lane)
  if (lane < TM) {
    const int m = lane;
    float mu = 0.f, s2 = 0.f;
    for (int d = 0; d < 64; ++d) { float x = wTmp[m * 64 + d]; mu += x; s2 += x * x; }
    mu *= (1.f / 64.f);
    s2 = s2 * (1.f / 64.f) - mu * mu;
    float inv = rsqrtf(s2 + 1e-5f);
    for (int d = 0; d < 64; ++d)
      wComb[(TM + m) * 64 + d] =
          (half_t)(((wTmp[m * 64 + d] - mu) * inv) * sL1g[d] + sL1b[d]);
  }
  WAVE_SYNC();

  // ---- stage 3: qkv = combined @ in_proj^T + b (32x64x192) ----
  // nt-outer / both-token-inner: each B fragment pair feeds 4 WMMAs.
  {
    const int col = lane & 15, sel = lane >> 4;
    v16h a00 = load_a(wComb,           64, 0,  lane);
    v16h a01 = load_a(wComb,           64, 32, lane);
    v16h a10 = load_a(wComb + TM * 64, 64, 0,  lane);
    v16h a11 = load_a(wComb + TM * 64, 64, 32, lane);
    float bias[12];
#pragma unroll
    for (int nt = 0; nt < 12; ++nt) bias[nt] = sBin[nt * 16 + col];
    v16h b0 = load_b(sWin, 64, 0,  lane);
    v16h b1 = load_b(sWin, 64, 32, lane);
#pragma unroll
    for (int nt = 0; nt < 12; ++nt) {
      const int n0 = nt * 16;
      v16h nb0, nb1;
      if (nt < 11) {
        nb0 = load_b(sWin + (n0 + 16) * 64, 64, 0,  lane);
        nb1 = load_b(sWin + (n0 + 16) * 64, 64, 32, lane);
      }
      v8f c0 = {}; c0 = wmma_f16(a00, b0, c0); c0 = wmma_f16(a01, b1, c0);
      v8f c1 = {}; c1 = wmma_f16(a10, b0, c1); c1 = wmma_f16(a11, b1, c1);
#pragma unroll
      for (int r = 0; r < 8; ++r) {
        const int row = r + sel * 8;
        wQKV[row * 192 + n0 + col]        = (half_t)(c0[r] + bias[nt]);
        wQKV[(TM + row) * 192 + n0 + col] = (half_t)(c1[r] + bias[nt]);
      }
      b0 = nb0; b1 = nb1;
    }
  }
  WAVE_SYNC();

  // ---- stage 4: 2-token / 4-head attention (2 (m,h) jobs per lane) ----
#pragma unroll
  for (int i = 0; i < 2; ++i) {
    const int job = lane * 2 + i;            // 0..63
    const int m = job >> 2, h = job & 3, h16 = h * 16;
    const half_t* r0 = wQKV + m * 192;        // token 0
    const half_t* r1 = wQKV + (TM + m) * 192; // token 1
    float s00 = 0, s01 = 0, s10 = 0, s11 = 0;
    for (int d = 0; d < 16; ++d) {
      float q0 = (float)r0[h16 + d],      q1 = (float)r1[h16 + d];
      float k0 = (float)r0[64 + h16 + d], k1 = (float)r1[64 + h16 + d];
      s00 += q0 * k0; s01 += q0 * k1; s10 += q1 * k0; s11 += q1 * k1;
    }
    s00 *= 0.25f; s01 *= 0.25f; s10 *= 0.25f; s11 *= 0.25f;
    float m0 = fmaxf(s00, s01), m1 = fmaxf(s10, s11);
    float e00 = __expf(s00 - m0), e01 = __expf(s01 - m0);
    float e10 = __expf(s10 - m1), e11 = __expf(s11 - m1);
    float i0 = 1.f / (e00 + e01), i1 = 1.f / (e10 + e11);
    float a00 = e00 * i0, a01 = e01 * i0, a10 = e10 * i1, a11 = e11 * i1;
    for (int d = 0; d < 16; ++d) {
      float v0 = (float)r0[128 + h16 + d], v1 = (float)r1[128 + h16 + d];
      wCtx[m * 64 + h16 + d]        = (half_t)(a00 * v0 + a01 * v1);
      wCtx[(TM + m) * 64 + h16 + d] = (half_t)(a10 * v0 + a11 * v1);
    }
  }
  WAVE_SYNC();

  // ---- stage 5: out_proj (32x64x64) + bias + residual -> f32 ----
  {
    const int col = lane & 15, sel = lane >> 4;
    v16h a00 = load_a(wCtx,           64, 0,  lane);
    v16h a01 = load_a(wCtx,           64, 32, lane);
    v16h a10 = load_a(wCtx + TM * 64, 64, 0,  lane);
    v16h a11 = load_a(wCtx + TM * 64, 64, 32, lane);
    float bias[4];
#pragma unroll
    for (int nt = 0; nt < 4; ++nt) bias[nt] = sBout[nt * 16 + col];
    v16h b0 = load_b(sWout, 64, 0,  lane);
    v16h b1 = load_b(sWout, 64, 32, lane);
#pragma unroll
    for (int nt = 0; nt < 4; ++nt) {
      const int n0 = nt * 16;
      v16h nb0, nb1;
      if (nt < 3) {
        nb0 = load_b(sWout + (n0 + 16) * 64, 64, 0,  lane);
        nb1 = load_b(sWout + (n0 + 16) * 64, 64, 32, lane);
      }
      v8f c0 = {}; c0 = wmma_f16(a00, b0, c0); c0 = wmma_f16(a01, b1, c0);
      v8f c1 = {}; c1 = wmma_f16(a10, b0, c1); c1 = wmma_f16(a11, b1, c1);
#pragma unroll
      for (int r = 0; r < 8; ++r) {
        const int row0 = r + sel * 8, row1 = TM + row0;
        wXf[row0 * 64 + n0 + col] =
            c0[r] + bias[nt] + (float)wComb[row0 * 64 + n0 + col];
        wXf[row1 * 64 + n0 + col] =
            c1[r] + bias[nt] + (float)wComb[row1 * 64 + n0 + col];
      }
      b0 = nb0; b1 = nb1;
    }
  }
  WAVE_SYNC();
  // LayerNorm2: one token-row per lane (32 rows)
  {
    const int r = lane;
    float mu = 0.f, s2 = 0.f;
    for (int d = 0; d < 64; ++d) { float x = wXf[r * 64 + d]; mu += x; s2 += x * x; }
    mu *= (1.f / 64.f);
    s2 = s2 * (1.f / 64.f) - mu * mu;
    float inv = rsqrtf(s2 + 1e-5f);
    for (int d = 0; d < 64; ++d)
      wXf[r * 64 + d] = ((wXf[r * 64 + d] - mu) * inv) * sL2g[d] + sL2b[d];
  }
  WAVE_SYNC();
  // mean-pool over the 2 tokens -> f16 A matrix for the head
  for (int e = lane; e < TM * 64; e += 32) {
    int m = e >> 6, d = e & 63;
    wPool[m * 64 + d] =
        (half_t)(0.5f * (wXf[m * 64 + d] + wXf[(TM + m) * 64 + d]));
  }
  WAVE_SYNC();

  // ---- stage 6: head: relu(pooled @ W3^T + b3) (16x64x32) ----
  {
    v16h a0 = load_a(wPool, 64, 0,  lane);
    v16h a1 = load_a(wPool, 64, 32, lane);
    const int col = lane & 15, sel = lane >> 4;
    const float bias0 = sB3[col], bias1 = sB3[16 + col];
    v16h b0 = load_b(sW3,           64, 0,  lane);
    v16h b1 = load_b(sW3,           64, 32, lane);
    v16h d0 = load_b(sW3 + 16 * 64, 64, 0,  lane);
    v16h d1 = load_b(sW3 + 16 * 64, 64, 32, lane);
    v8f c0 = {}; c0 = wmma_f16(a0, b0, c0); c0 = wmma_f16(a1, b1, c0);
    v8f c1 = {}; c1 = wmma_f16(a0, d0, c1); c1 = wmma_f16(a1, d1, c1);
#pragma unroll
    for (int r = 0; r < 8; ++r) {
      const int row = r + sel * 8;
      wHead[row * 32 + col]      = fmaxf(c0[r] + bias0, 0.0f);
      wHead[row * 32 + 16 + col] = fmaxf(c1[r] + bias1, 0.0f);
    }
  }
  WAVE_SYNC();
  // final 32-wide dot with W4 + b4 (one sample per lane<16)
  if (lane < TM) {
    const int m = lane;
    float acc = sB4[0];
    for (int j = 0; j < 32; ++j) acc += wHead[m * 32 + j] * sW4[j];
    const int s = g0 + m;
    if (s < B) out[s] = acc;
  }
}

extern "C" void kernel_launch(void* const* d_in, const int* in_sizes, int n_in,
                              void* d_out, int out_size, void* d_ws, size_t ws_size,
                              hipStream_t stream) {
  (void)n_in; (void)d_ws; (void)ws_size; (void)out_size;
  const int*   hw_idx   = (const int*)d_in[0];
  const int*   op_idx   = (const int*)d_in[1];
  const int*   wd_idx   = (const int*)d_in[2];
  const float* hw_embed = (const float*)d_in[3];
  const float* W1   = (const float*)d_in[4];
  const float* b1   = (const float*)d_in[5];
  const float* W2   = (const float*)d_in[6];
  const float* b2   = (const float*)d_in[7];
  const float* ln1g = (const float*)d_in[8];
  const float* ln1b = (const float*)d_in[9];
  const float* Win  = (const float*)d_in[10];
  const float* bin  = (const float*)d_in[11];
  const float* Wout = (const float*)d_in[12];
  const float* bout = (const float*)d_in[13];
  const float* ln2g = (const float*)d_in[14];
  const float* ln2b = (const float*)d_in[15];
  const float* W3   = (const float*)d_in[16];
  const float* b3   = (const float*)d_in[17];
  const float* W4   = (const float*)d_in[18];
  const float* b4   = (const float*)d_in[19];
  float* out = (float*)d_out;

  const int B = in_sizes[0];
  const int samples_per_block = NWAVES * TM;   // 128
  const int grid = (B + samples_per_block - 1) / samples_per_block;
  mhlp_predictor_kernel<<<grid, NTHREADS, SMEM_TOTAL, stream>>>(
      hw_idx, op_idx, wd_idx, hw_embed, W1, b1, W2, b2, ln1g, ln1b,
      Win, bin, Wout, bout, ln2g, ln2b, W3, b3, W4, b4, out, B);
}
